// MultiheadAttention_31602369364652
// MI455X (gfx1250) — matrix operand order repaired and checked by reading, not compiled, not run on hardware
//
#include <hip/hip_runtime.h>
#include <hip/hip_bf16.h>

// ---------------------------------------------------------------------------
// MI455X (gfx1250) multi-head attention with RoPE.
// B=2, L=2048, D=1024, H=16, hd=64.  ~70 GFLOP vs ~100MB traffic -> compute
// bound -> bf16 WMMA (v_wmma_f32_16x16x32_bf16) for all matrix work.
// GEMMs stage A/B K-slices in LDS with double-buffered async global->LDS
// copies (GLOBAL_LOAD_ASYNC_TO_LDS_B128 / s_wait_asynccnt).
// ---------------------------------------------------------------------------

#define D_MODEL 1024
#define L_SEQ   2048
#define N_HEAD  16
#define HEAD_D  64
#define N_BATCH 2
#define M_ROWS  (N_BATCH * L_SEQ)   // 4096
#define LDS_PITCH 40                // 32 data + 8 pad bf16 per row (16B align)

typedef __attribute__((ext_vector_type(16))) __bf16 v16bf;
typedef __attribute__((ext_vector_type(8)))  float  v8f;

union FragU { uint4 q[2]; v16bf v; };   // 32 bytes: 16 bf16 values

__device__ __forceinline__ unsigned short f2bf(float f) {
    unsigned u = __float_as_uint(f);
    unsigned r = u + 0x7FFFu + ((u >> 16) & 1u);   // round-to-nearest-even
    return (unsigned short)(r >> 16);
}
__device__ __forceinline__ float bf2f(unsigned short h) {
    return __uint_as_float(((unsigned)h) << 16);
}
__device__ __forceinline__ v8f vzero8() {
    v8f z = {0.f, 0.f, 0.f, 0.f, 0.f, 0.f, 0.f, 0.f};
    return z;
}

// Async global->LDS 16B copy.  ldsOff = wave-relative LDS byte offset
// (low 32 bits of the generic shared-memory address), gaddr = global address.
__device__ __forceinline__ void async_copy_b128(unsigned ldsOff,
                                                unsigned long long gaddr) {
    asm volatile("global_load_async_to_lds_b128 %0, %1, off"
                 :: "v"(ldsOff), "v"(gaddr) : "memory");
}

// ---------------------------------------------------------------------------
// f32 -> bf16 conversion
// ---------------------------------------------------------------------------
__global__ void cvt_f32_bf16(const float* __restrict__ in,
                             unsigned short* __restrict__ out, int n) {
    int i = blockIdx.x * blockDim.x + threadIdx.x;
    if (i < n) out[i] = f2bf(in[i]);
}

// ---------------------------------------------------------------------------
// GEMM: out[m,n] = sum_k A[m,k] * W[n,k]   (A: M x K bf16, W: N x K bf16)
// M=4096, N=1024, K=1024.  Block tile 128x128, 8 waves, each wave 32x64.
// K-slices (128x32) of A and B staged in LDS with double-buffered async
// copies; 8 WMMAs per wave per 32-K step.
// mode 0: store bf16 to [b,h,l,hd]      (Q, K projections)
// mode 1: store bf16 to [b,h,hd,l]      (V projection, transposed)
// mode 2: store f32  row-major          (final output projection)
// ---------------------------------------------------------------------------
__global__ __launch_bounds__(256)
void gemm_bf16_wmma(const unsigned short* __restrict__ A,
                    const unsigned short* __restrict__ W,
                    float* __restrict__ outF,
                    unsigned short* __restrict__ outB,
                    int mode) {
    __shared__ __align__(16) unsigned short ldsA[2][128 * LDS_PITCH];
    __shared__ __align__(16) unsigned short ldsB[2][128 * LDS_PITCH];

    const int K = D_MODEL;
    const int tid  = threadIdx.x;
    const int w    = tid >> 5;
    const int lane = tid & 31;
    const int half = lane >> 4;
    const int ln   = lane & 15;
    const int wm = w >> 1;           // 0..3
    const int wn = w & 1;            // 0..1
    const int rowBase0 = blockIdx.y * 128;
    const int colBase0 = blockIdx.x * 128;

    // Each thread copies two 16B chunks of A and two of B per stage.
    // chunk c (0..511): row = c>>2, colOff = (c&3)*8 elements.
    const int c0  = tid * 2;
    const int cr0 = c0 >> 2,        cc0 = (c0 & 3) * 8;
    const int cr1 = (c0 + 1) >> 2,  cc1 = ((c0 + 1) & 3) * 8;

    auto issue_stage = [&](int kb, int buf) {
        unsigned la0 = (unsigned)(uintptr_t)&ldsA[buf][cr0 * LDS_PITCH + cc0];
        unsigned la1 = (unsigned)(uintptr_t)&ldsA[buf][cr1 * LDS_PITCH + cc1];
        unsigned lb0 = (unsigned)(uintptr_t)&ldsB[buf][cr0 * LDS_PITCH + cc0];
        unsigned lb1 = (unsigned)(uintptr_t)&ldsB[buf][cr1 * LDS_PITCH + cc1];
        async_copy_b128(la0, (unsigned long long)(uintptr_t)
                        (A + (size_t)(rowBase0 + cr0) * K + kb + cc0));
        async_copy_b128(la1, (unsigned long long)(uintptr_t)
                        (A + (size_t)(rowBase0 + cr1) * K + kb + cc1));
        async_copy_b128(lb0, (unsigned long long)(uintptr_t)
                        (W + (size_t)(colBase0 + cr0) * K + kb + cc0));
        async_copy_b128(lb1, (unsigned long long)(uintptr_t)
                        (W + (size_t)(colBase0 + cr1) * K + kb + cc1));
    };

    v8f acc[2][4];
#pragma unroll
    for (int i = 0; i < 2; ++i)
#pragma unroll
        for (int j = 0; j < 4; ++j) acc[i][j] = vzero8();

    issue_stage(0, 0);

    int buf = 0;
    for (int kb = 0; kb < K; kb += 32) {
        if (kb + 32 < K) {
            issue_stage(kb + 32, buf ^ 1);
            // retire current stage's 4 copies; next stage's 4 stay in flight
            asm volatile("s_wait_asynccnt 0x4" ::: "memory");
        } else {
            asm volatile("s_wait_asynccnt 0x0" ::: "memory");
        }
        __syncthreads();

        FragU a[2];
#pragma unroll
        for (int i = 0; i < 2; ++i) {
            const unsigned short* p =
                &ldsA[buf][(wm * 32 + i * 16 + ln) * LDS_PITCH];
            a[i].q[0] = *(const uint4*)(p + half * 8);
            a[i].q[1] = *(const uint4*)(p + 16 + half * 8);
        }
        FragU b[4];
#pragma unroll
        for (int j = 0; j < 4; ++j) {
            const unsigned short* p =
                &ldsB[buf][(wn * 64 + j * 16 + ln) * LDS_PITCH + half * 8];
            b[j].q[0] = *(const uint4*)(p);
            b[j].q[1] = *(const uint4*)(p + 16);
        }
#pragma unroll
        for (int i = 0; i < 2; ++i)
#pragma unroll
            for (int j = 0; j < 4; ++j)
                acc[i][j] = __builtin_amdgcn_wmma_f32_16x16x32_bf16(
                    false, a[i].v, false, b[j].v, (short)0, acc[i][j],
                    false, false);

        __syncthreads();   // nobody may still read buf when it is refilled
        buf ^= 1;
    }

#pragma unroll
    for (int i = 0; i < 2; ++i)
#pragma unroll
        for (int j = 0; j < 4; ++j)
#pragma unroll
            for (int r = 0; r < 8; ++r) {
                int row = rowBase0 + wm * 32 + i * 16 + half * 8 + r;
                int col = colBase0 + wn * 64 + j * 16 + ln;
                float v = acc[i][j][r];
                if (mode == 2) {
                    outF[(size_t)row * D_MODEL + col] = v;
                } else {
                    int b_ = row >> 11;            // row / L_SEQ
                    int l  = row & (L_SEQ - 1);
                    int h  = col >> 6;             // col / HEAD_D
                    int d  = col & (HEAD_D - 1);
                    size_t off;
                    if (mode == 0)
                        off = (((size_t)(b_ * N_HEAD + h) * L_SEQ) + l) * HEAD_D + d;
                    else
                        off = (((size_t)(b_ * N_HEAD + h) * HEAD_D) + d) * L_SEQ + l;
                    outB[off] = f2bf(v);
                }
            }
}

// ---------------------------------------------------------------------------
// RoPE in place on Q and K, layout [B,H,L,hd].  One thread per rotation pair.
// ---------------------------------------------------------------------------
__global__ void rope_kernel(unsigned short* __restrict__ q,
                            unsigned short* __restrict__ k,
                            const float* __restrict__ cosT,
                            const float* __restrict__ sinT) {
    int idx = blockIdx.x * blockDim.x + threadIdx.x;  // B*H*L*32 threads
    int d  = idx & 31;
    int l  = (idx >> 5) & (L_SEQ - 1);
    int bh = idx >> 16;
    size_t base = ((size_t)bh * L_SEQ + l) * HEAD_D;
    float c1 = cosT[l * HEAD_D + d],      s1 = sinT[l * HEAD_D + d];
    float c2 = cosT[l * HEAD_D + d + 32], s2 = sinT[l * HEAD_D + d + 32];

    float x1 = bf2f(q[base + d]), x2 = bf2f(q[base + d + 32]);
    q[base + d]      = f2bf(x1 * c1 - x2 * s1);
    q[base + d + 32] = f2bf(x2 * c2 + x1 * s2);

    float y1 = bf2f(k[base + d]), y2 = bf2f(k[base + d + 32]);
    k[base + d]      = f2bf(y1 * c1 - y2 * s1);
    k[base + d + 32] = f2bf(y2 * c2 + y1 * s2);
}

// ---------------------------------------------------------------------------
// Causal flash attention.  Q,K: [B,H,L,hd] bf16, V: [B,H,hd,L] bf16.
// Grid: (L/128, B*H). 256 threads = 8 waves; wave w owns 16 query rows.
// Online softmax; P goes through per-wave LDS to reach WMMA A-frag layout.
// Output: bf16 [B,L,D] (row=b*L+l, col=h*64+d) for the final projection.
// ---------------------------------------------------------------------------
__global__ __launch_bounds__(256)
void flash_attn(const unsigned short* __restrict__ Q,
                const unsigned short* __restrict__ Km,
                const unsigned short* __restrict__ Vt,
                unsigned short* __restrict__ out) {
    __shared__ __align__(16) unsigned short ldsP[8][16][40];  // +8 pad / row

    const int tid  = threadIdx.x;
    const int w    = tid >> 5;
    const int lane = tid & 31;
    const int half = lane >> 4;
    const int ln   = lane & 15;
    const int bh   = blockIdx.y;          // b*N_HEAD + h
    const int b_   = bh >> 4;
    const int h    = bh & 15;
    const int qBase = blockIdx.x * 128 + w * 16;

    const unsigned short* Qp = Q  + (size_t)bh * L_SEQ * HEAD_D;
    const unsigned short* Kp = Km + (size_t)bh * L_SEQ * HEAD_D;
    const unsigned short* Vp = Vt + (size_t)bh * HEAD_D * L_SEQ;

    // Q A-fragments for this wave's 16 rows, 2 k-steps over hd=64.
    FragU qf[2];
    {
        const unsigned short* p = Qp + (size_t)(qBase + ln) * HEAD_D;
#pragma unroll
        for (int kk = 0; kk < 2; ++kk) {
            qf[kk].q[0] = *(const uint4*)(p + kk * 32 + half * 8);
            qf[kk].q[1] = *(const uint4*)(p + kk * 32 + 16 + half * 8);
        }
    }

    v8f o[4];
#pragma unroll
    for (int j = 0; j < 4; ++j) o[j] = vzero8();
    float mrow[8], lrow[8];
#pragma unroll
    for (int r = 0; r < 8; ++r) { mrow[r] = -3.0e38f; lrow[r] = 0.f; }

    const float scl = 0.125f;  // 1/sqrt(hd)

    for (int kb = 0; kb < qBase + 16; kb += 32) {
        // prefetch next key tile rows (speculative; dropped if OOB)
        __builtin_prefetch(Kp + (size_t)(kb + 32 + ln) * HEAD_D, 0, 0);
        __builtin_prefetch(Vp + (size_t)(ln * 4) * L_SEQ + kb + 32, 0, 0);

        // ---- S = Q K^T for 32 keys (two 16-col subtiles) ----
        v8f s[2];
        s[0] = vzero8(); s[1] = vzero8();
#pragma unroll
        for (int sub = 0; sub < 2; ++sub) {
            const unsigned short* p = Kp + (size_t)(kb + sub * 16 + ln) * HEAD_D;
#pragma unroll
            for (int kk = 0; kk < 2; ++kk) {
                FragU kf;
                kf.q[0] = *(const uint4*)(p + kk * 32 + half * 8);
                kf.q[1] = *(const uint4*)(p + kk * 32 + 16 + half * 8);
                s[sub] = __builtin_amdgcn_wmma_f32_16x16x32_bf16(
                    false, qf[kk].v, false, kf.v, (short)0, s[sub],
                    false, false);
            }
        }

        // ---- scale + causal mask + per-row tile max ----
        float pmax[8];
#pragma unroll
        for (int r = 0; r < 8; ++r) {
            int qrow = qBase + half * 8 + r;
            int key0 = kb + ln, key1 = kb + 16 + ln;
            float v0 = (key0 <= qrow) ? s[0][r] * scl : -3.0e38f;
            float v1 = (key1 <= qrow) ? s[1][r] * scl : -3.0e38f;
            s[0][r] = v0; s[1][r] = v1;
            pmax[r] = fmaxf(v0, v1);
        }
#pragma unroll
        for (int m = 1; m < 16; m <<= 1)
#pragma unroll
            for (int r = 0; r < 8; ++r)
                pmax[r] = fmaxf(pmax[r], __shfl_xor(pmax[r], m, 32));

        // ---- online softmax update ----
        float sc[8], rsum[8];
#pragma unroll
        for (int r = 0; r < 8; ++r) {
            float newm = fmaxf(mrow[r], pmax[r]);
            sc[r] = __expf(mrow[r] - newm);
            float p0 = __expf(s[0][r] - newm);
            float p1 = __expf(s[1][r] - newm);
            s[0][r] = p0; s[1][r] = p1;
            rsum[r] = p0 + p1;
            mrow[r] = newm;
        }
#pragma unroll
        for (int m = 1; m < 16; m <<= 1)
#pragma unroll
            for (int r = 0; r < 8; ++r)
                rsum[r] += __shfl_xor(rsum[r], m, 32);
#pragma unroll
        for (int r = 0; r < 8; ++r) {
            lrow[r] = lrow[r] * sc[r] + rsum[r];
#pragma unroll
            for (int j = 0; j < 4; ++j) o[j][r] *= sc[r];
        }

        // ---- P (16x32) through per-wave LDS into A-fragment layout ----
#pragma unroll
        for (int r = 0; r < 8; ++r) {
            int m_ = half * 8 + r;
            ldsP[w][m_][ln]      = f2bf(s[0][r]);
            ldsP[w][m_][16 + ln] = f2bf(s[1][r]);
        }
        FragU pf;
        pf.q[0] = *(const uint4*)&ldsP[w][ln][half * 8];
        pf.q[1] = *(const uint4*)&ldsP[w][ln][16 + half * 8];

        // ---- O += P * V  (V^T rows are contiguous over keys) ----
#pragma unroll
        for (int j = 0; j < 4; ++j) {
            const unsigned short* p =
                Vp + (size_t)(j * 16 + ln) * L_SEQ + kb + half * 8;
            FragU vf;
            vf.q[0] = *(const uint4*)(p);
            vf.q[1] = *(const uint4*)(p + 16);
            o[j] = __builtin_amdgcn_wmma_f32_16x16x32_bf16(
                false, pf.v, false, vf.v, (short)0, o[j], false, false);
        }
    }

    // ---- normalize and write bf16 [B, L, H*hd] ----
#pragma unroll
    for (int r = 0; r < 8; ++r) {
        float inv = 1.f / lrow[r];
        int qrow = qBase + half * 8 + r;
        size_t rowOff = ((size_t)b_ * L_SEQ + qrow) * D_MODEL + h * HEAD_D;
#pragma unroll
        for (int j = 0; j < 4; ++j)
            out[rowOff + j * 16 + ln] = f2bf(o[j][r] * inv);
    }
}

// ---------------------------------------------------------------------------
// Host-side launch
// ---------------------------------------------------------------------------
extern "C" void kernel_launch(void* const* d_in, const int* in_sizes, int n_in,
                              void* d_out, int out_size, void* d_ws,
                              size_t ws_size, hipStream_t stream) {
    const float* x    = (const float*)d_in[0];
    const float* cosT = (const float*)d_in[1];
    const float* sinT = (const float*)d_in[2];
    const float* wq   = (const float*)d_in[3];
    const float* wk   = (const float*)d_in[4];
    const float* wv   = (const float*)d_in[5];
    const float* wo   = (const float*)d_in[6];
    float* out = (float*)d_out;

    const size_t nX = (size_t)M_ROWS * D_MODEL;     // 4M elements
    const size_t nW = (size_t)D_MODEL * D_MODEL;    // 1M elements

    unsigned short* xb  = (unsigned short*)d_ws;
    unsigned short* wqb = xb  + nX;
    unsigned short* wkb = wqb + nW;
    unsigned short* wvb = wkb + nW;
    unsigned short* wob = wvb + nW;
    unsigned short* qb  = wob + nW;   // [B,H,L,hd]
    unsigned short* kb  = qb  + nX;   // [B,H,L,hd]
    unsigned short* vT  = kb  + nX;   // [B,H,hd,L]
    unsigned short* ao  = vT  + nX;   // [B,L,D]
    // total workspace: 48 MiB of bf16

    cvt_f32_bf16<<<(int)(nX / 256), 256, 0, stream>>>(x,  xb,  (int)nX);
    cvt_f32_bf16<<<(int)(nW / 256), 256, 0, stream>>>(wq, wqb, (int)nW);
    cvt_f32_bf16<<<(int)(nW / 256), 256, 0, stream>>>(wk, wkb, (int)nW);
    cvt_f32_bf16<<<(int)(nW / 256), 256, 0, stream>>>(wv, wvb, (int)nW);
    cvt_f32_bf16<<<(int)(nW / 256), 256, 0, stream>>>(wo, wob, (int)nW);

    dim3 gg(D_MODEL / 128, M_ROWS / 128);   // (8, 32)
    gemm_bf16_wmma<<<gg, 256, 0, stream>>>(xb, wqb, nullptr, qb, 0);
    gemm_bf16_wmma<<<gg, 256, 0, stream>>>(xb, wkb, nullptr, kb, 0);
    gemm_bf16_wmma<<<gg, 256, 0, stream>>>(xb, wvb, nullptr, vT, 1);

    // B*H*L*32 rotation pairs = 2,097,152 threads
    rope_kernel<<<8192, 256, 0, stream>>>(qb, kb, cosT, sinT);

    dim3 ga(L_SEQ / 128, N_BATCH * N_HEAD);  // (16, 32)
    flash_attn<<<ga, 256, 0, stream>>>(qb, kb, vT, ao);

    gemm_bf16_wmma<<<gg, 256, 0, stream>>>(ao, wob, out, nullptr, 2);
}
